// GCN_85899345920723
// MI455X (gfx1250) — compile-verified, matrix operand order checked by reading, not run on hardware
//
#include <hip/hip_runtime.h>
#include <hip/hip_bf16.h>

#define NN 20000
#define NE 160000

typedef _Float16 h8  __attribute__((ext_vector_type(8)));
typedef _Float16 h16 __attribute__((ext_vector_type(16)));
typedef float    f8  __attribute__((ext_vector_type(8)));

// ---------------- utility kernels ----------------

__global__ void gcn_zero_f32(float* __restrict__ p, size_t n) {
    size_t i = (size_t)blockIdx.x * blockDim.x + threadIdx.x;
    if (i < n) p[i] = 0.0f;
}

__global__ void gcn_deg(const int* __restrict__ dst, float* __restrict__ deg) {
    int e = blockIdx.x * blockDim.x + threadIdx.x;
    if (e < NE) atomicAdd(&deg[dst[e]], 1.0f);
}

__global__ void gcn_dinv(const float* __restrict__ deg, float* __restrict__ dinv) {
    int n = blockIdx.x * blockDim.x + threadIdx.x;
    if (n < NN) dinv[n] = rsqrtf(deg[n] + 1.0f);   // +1 for self-loop
}

__global__ void gcn_enorm(const int* __restrict__ src, const int* __restrict__ dst,
                          const float* __restrict__ dinv, float* __restrict__ enorm) {
    int e = blockIdx.x * blockDim.x + threadIdx.x;
    if (e < NE) enorm[e] = dinv[src[e]] * dinv[dst[e]];
}

__global__ void gcn_cvt_f32_f16(const float* __restrict__ in, _Float16* __restrict__ out, size_t n) {
    size_t i = (size_t)blockIdx.x * blockDim.x + threadIdx.x;
    if (i < n) out[i] = (_Float16)in[i];
}

// W (f32, fin x fout row-major) -> Wt (f16, fout x fin row-major) i.e. transposed
__global__ void gcn_wconv(const float* __restrict__ W, _Float16* __restrict__ Wt,
                          int fin, int fout) {
    size_t i = (size_t)blockIdx.x * blockDim.x + threadIdx.x;
    if (i < (size_t)fin * fout) {
        int k = (int)(i / fout);
        int c = (int)(i % fout);
        Wt[(size_t)c * fin + k] = (_Float16)W[i];
    }
}

// -------- fragment loaders (ISA 16-bit 16x32 layout) --------
// lane L holds row/col = L%16, Ks {kb..kb+7, kb+16..kb+23}, kb = (L/16)*8
__device__ __forceinline__ h16 load_frag(const _Float16* p) {
    h8 lo = *(const h8*)p;
    h8 hi = *(const h8*)(p + 16);
    return __builtin_shufflevector(lo, hi, 0,1,2,3,4,5,6,7,8,9,10,11,12,13,14,15);
}

__device__ __forceinline__ f8 wmma_f16(h16 a, h16 b, f8 c) {
    return __builtin_amdgcn_wmma_f32_16x16x32_f16(false, a, false, b, (short)0, c, false, false);
}

__device__ __forceinline__ void store_tile_f16(f8 c, _Float16* __restrict__ Mout,
                                               int row0, int col0, int fout,
                                               int lane) {
    const int r16  = lane & 15;
    const int mrow = row0 + ((lane >> 4) << 3);
    _Float16* op = Mout + (size_t)mrow * fout + col0 + r16;
#pragma unroll
    for (int r = 0; r < 8; ++r)
        op[(size_t)r * fout] = (_Float16)c[r];
}

// ---------------- WMMA GEMM, 32x64 register tile per wave (fout % 64 == 0) ----------------
// block = 128 threads = 4 waves; block tile = 32 rows x 256 cols; wave w -> cols [w*64, w*64+64)
// Per K=32 step: 2 A-frags + 4 B-frags -> 8 WMMAs (~21.3 FLOP/byte from L1/L2).
__global__ __launch_bounds__(128) void gcn_gemm_wmma_32x64(
    const _Float16* __restrict__ A, const _Float16* __restrict__ Bt,
    _Float16* __restrict__ Mout, int fin, int fout)
{
    const int lane  = threadIdx.x & 31;
    const int wave  = threadIdx.x >> 5;
    const int wcol0 = blockIdx.x * 256 + wave * 64;
    if (wcol0 >= fout) return;                       // wave-uniform guard
    const int row0  = blockIdx.y * 32;
    const int r16   = lane & 15;
    const int kb    = (lane >> 4) << 3;              // 0 or 8

    const _Float16* ap0 = A + (size_t)(row0 + r16) * fin + kb;
    const _Float16* ap1 = ap0 + (size_t)16 * fin;
    const _Float16* bp[4];
#pragma unroll
    for (int j = 0; j < 4; ++j)
        bp[j] = Bt + (size_t)(wcol0 + 16 * j + r16) * fin + kb;

    f8 acc0[4], acc1[4];
#pragma unroll
    for (int j = 0; j < 4; ++j) {
        acc0[j] = (f8){0.f,0.f,0.f,0.f,0.f,0.f,0.f,0.f};
        acc1[j] = acc0[j];
    }

    for (int k = 0; k < fin; k += 32) {
        h16 a0 = load_frag(ap0 + k);
        h16 a1 = load_frag(ap1 + k);
        h16 bf[4];
#pragma unroll
        for (int j = 0; j < 4; ++j) bf[j] = load_frag(bp[j] + k);
#pragma unroll
        for (int j = 0; j < 4; ++j) acc0[j] = wmma_f16(a0, bf[j], acc0[j]);
#pragma unroll
        for (int j = 0; j < 4; ++j) acc1[j] = wmma_f16(a1, bf[j], acc1[j]);
    }

#pragma unroll
    for (int j = 0; j < 4; ++j) {
        store_tile_f16(acc0[j], Mout, row0,      wcol0 + 16 * j, fout, lane);
        store_tile_f16(acc1[j], Mout, row0 + 16, wcol0 + 16 * j, fout, lane);
    }
}

// ---------------- WMMA GEMM, 32x32 register tile per wave (fout % 32 == 0) ----------------
__global__ __launch_bounds__(128) void gcn_gemm_wmma_32x32(
    const _Float16* __restrict__ A, const _Float16* __restrict__ Bt,
    _Float16* __restrict__ Mout, int fin, int fout)
{
    const int lane  = threadIdx.x & 31;
    const int wave  = threadIdx.x >> 5;
    const int wcol0 = blockIdx.x * 128 + wave * 32;
    if (wcol0 >= fout) return;                       // wave-uniform guard
    const int row0  = blockIdx.y * 32;
    const int r16   = lane & 15;
    const int kb    = (lane >> 4) << 3;

    const _Float16* ap0 = A  + (size_t)(row0 + r16) * fin + kb;
    const _Float16* ap1 = ap0 + (size_t)16 * fin;
    const _Float16* bp0 = Bt + (size_t)(wcol0 + r16) * fin + kb;
    const _Float16* bp1 = bp0 + (size_t)16 * fin;

    f8 c00 = {0.f,0.f,0.f,0.f,0.f,0.f,0.f,0.f};
    f8 c01 = c00, c10 = c00, c11 = c00;

    for (int k = 0; k < fin; k += 32) {
        h16 a0 = load_frag(ap0 + k);
        h16 a1 = load_frag(ap1 + k);
        h16 b0 = load_frag(bp0 + k);
        h16 b1 = load_frag(bp1 + k);
        c00 = wmma_f16(a0, b0, c00);
        c01 = wmma_f16(a0, b1, c01);
        c10 = wmma_f16(a1, b0, c10);
        c11 = wmma_f16(a1, b1, c11);
    }

    store_tile_f16(c00, Mout, row0,      wcol0,      fout, lane);
    store_tile_f16(c01, Mout, row0,      wcol0 + 16, fout, lane);
    store_tile_f16(c10, Mout, row0 + 16, wcol0,      fout, lane);
    store_tile_f16(c11, Mout, row0 + 16, wcol0 + 16, fout, lane);
}

// ---------------- WMMA GEMM, simple 16x16 per wave (fallback, fout % 16 == 0) ------------
__global__ __launch_bounds__(128) void gcn_gemm_wmma_16(
    const _Float16* __restrict__ A, const _Float16* __restrict__ Bt,
    _Float16* __restrict__ Mout, int fin, int fout)
{
    const int lane = threadIdx.x & 31;
    const int wave = threadIdx.x >> 5;
    const int col0 = (blockIdx.x * 4 + wave) * 16;
    if (col0 >= fout) return;                        // wave-uniform guard
    const int row0 = blockIdx.y * 16;
    const int r16  = lane & 15;
    const int kb   = (lane >> 4) << 3;

    const _Float16* ap = A  + (size_t)(row0 + r16) * fin + kb;
    const _Float16* bp = Bt + (size_t)(col0 + r16) * fin + kb;

    f8 c = {0.f,0.f,0.f,0.f,0.f,0.f,0.f,0.f};
    for (int k = 0; k < fin; k += 32)
        c = wmma_f16(load_frag(ap + k), load_frag(bp + k), c);

    store_tile_f16(c, Mout, row0, col0, fout, lane);
}

// ---------------- edge scatter: agg[dst] += m[src] * norm  (one block per edge) ----------
// blockDim.x = fout/8; thread t handles features [8t, 8t+8)
__global__ void gcn_scatter(const int* __restrict__ src, const int* __restrict__ dst,
                            const float* __restrict__ enorm, const _Float16* __restrict__ Mh,
                            float* __restrict__ agg, int fout)
{
    const int e = blockIdx.x;
    const int s = src[e], d = dst[e];
    const float w = enorm[e];
    const int f0 = threadIdx.x * 8;
    h8 mv = *(const h8*)(Mh + (size_t)s * fout + f0);
    float* ap = agg + (size_t)d * fout + f0;
#pragma unroll
    for (int j = 0; j < 8; ++j)
        atomicAdd(ap + j, (float)mv[j] * w);
}

// ---------------- finalize: self-loop + bias + relu + f16 convert ----------------
__global__ void gcn_finalize(const float* __restrict__ agg, const _Float16* __restrict__ Mh,
                             const float* __restrict__ dinv, const float* __restrict__ bias,
                             _Float16* __restrict__ Hh, float* __restrict__ Out32,
                             int fout, int do_relu)
{
    const int n = blockIdx.y;
    const int f = blockIdx.x * blockDim.x + threadIdx.x;
    if (f >= fout) return;
    const size_t i = (size_t)n * fout + f;
    const float sn = dinv[n];
    float v = agg[i] + (float)Mh[i] * sn * sn + bias[f];
    if (do_relu) v = fmaxf(v, 0.0f);
    if (Out32) Out32[i] = v;
    else       Hh[i]    = (_Float16)v;
}

// ---------------- host ----------------

static const int GCN_DIMS[9] = {512, 2048, 1024, 512, 256, 128, 64, 32, 16};

extern "C" void kernel_launch(void* const* d_in, const int* in_sizes, int n_in,
                              void* d_out, int out_size, void* d_ws, size_t ws_size,
                              hipStream_t stream)
{
    const float* x   = (const float*)d_in[0];
    const int*   ei  = (const int*)d_in[1];
    const int*   src = ei;
    const int*   dst = ei + NE;

    // workspace layout (all regions 256B aligned)
    auto au = [](size_t v) { return (v + 255) & ~(size_t)255; };
    size_t wtot = 0;
    for (int i = 0; i < 8; ++i) wtot += (size_t)GCN_DIMS[i] * GCN_DIMS[i + 1];

    char* ws = (char*)d_ws;
    size_t off = 0;
    float*     deg   = (float*)(ws + off);     off += au((size_t)NN * 4);
    float*     dinv  = (float*)(ws + off);     off += au((size_t)NN * 4);
    float*     enorm = (float*)(ws + off);     off += au((size_t)NE * 4);
    _Float16*  Wt    = (_Float16*)(ws + off);  off += au(wtot * 2);
    _Float16*  Hh    = (_Float16*)(ws + off);  off += au((size_t)NN * 2048 * 2);
    _Float16*  Mh    = (_Float16*)(ws + off);  off += au((size_t)NN * 2048 * 2);
    float*     agg   = (float*)(ws + off);     off += au((size_t)NN * 2048 * 4);
    if (off > ws_size) return;  // scratch too small (deterministic no-op)

    // graph normalization
    gcn_zero_f32<<<(NN + 255) / 256, 256, 0, stream>>>(deg, (size_t)NN);
    gcn_deg     <<<(NE + 255) / 256, 256, 0, stream>>>(dst, deg);
    gcn_dinv    <<<(NN + 255) / 256, 256, 0, stream>>>(deg, dinv);
    gcn_enorm   <<<(NE + 255) / 256, 256, 0, stream>>>(src, dst, dinv, enorm);

    // x -> f16 activations
    {
        size_t n0 = (size_t)NN * GCN_DIMS[0];
        gcn_cvt_f32_f16<<<(unsigned)((n0 + 255) / 256), 256, 0, stream>>>(x, Hh, n0);
    }

    size_t woff = 0;
    for (int i = 0; i < 8; ++i) {
        const int fin  = GCN_DIMS[i];
        const int fout = GCN_DIMS[i + 1];
        const float* W = (const float*)d_in[2 + 2 * i];
        const float* b = (const float*)d_in[3 + 2 * i];
        _Float16* Wti  = Wt + woff;
        woff += (size_t)fin * fout;

        // W -> transposed f16
        size_t wn = (size_t)fin * fout;
        gcn_wconv<<<(unsigned)((wn + 255) / 256), 256, 0, stream>>>(W, Wti, fin, fout);

        // m = h @ W via WMMA
        if (fout % 64 == 0) {
            dim3 gg((fout + 255) / 256, NN / 32);
            gcn_gemm_wmma_32x64<<<gg, 128, 0, stream>>>(Hh, Wti, Mh, fin, fout);
        } else if (fout % 32 == 0) {
            dim3 gg((fout + 127) / 128, NN / 32);
            gcn_gemm_wmma_32x32<<<gg, 128, 0, stream>>>(Hh, Wti, Mh, fin, fout);
        } else {
            dim3 gg((fout + 63) / 64, NN / 16);
            gcn_gemm_wmma_16<<<gg, 128, 0, stream>>>(Hh, Wti, Mh, fin, fout);
        }

        // agg = segment_sum over edges
        size_t an = (size_t)NN * fout;
        gcn_zero_f32<<<(unsigned)((an + 255) / 256), 256, 0, stream>>>(agg, an);
        gcn_scatter<<<NE, fout / 8, 0, stream>>>(src, dst, enorm, Mh, agg, fout);

        // + self loop + bias (+ relu), convert for next layer
        dim3 gf((fout + 255) / 256, NN);
        gcn_finalize<<<gf, 256, 0, stream>>>(agg, Mh, dinv, b, Hh,
                                             (i == 7) ? (float*)d_out : nullptr,
                                             fout, (i < 7) ? 1 : 0);
    }
}
